// ViTConditionFusor_6854767805108
// MI455X (gfx1250) — compile-verified, matrix operand order checked by reading, not run on hardware
//
#include <hip/hip_runtime.h>

// ---------------------------------------------------------------------------
// CDNA5 (gfx1250) bf16 WMMA implementation of the ViT condition fusor.
// All heavy math -> v_wmma_f32_16x16x32_bf16. f32 kept for softmax/epilogues.
// k_gemm: 2Mx4N tiles, A double-buffered, B single-buffered (each B fragment
// feeds two WMMAs, so load waits amortize).  k_embed: 1Mx4N, fully
// double-buffered (register-pressure safe with the patch-gather temps).
// ---------------------------------------------------------------------------

typedef __attribute__((ext_vector_type(16))) __bf16 bf16x16;
typedef __attribute__((ext_vector_type(8)))  float  f32x8;

union Frag {
  bf16x16 bf;
  uint4 q[2];
  unsigned short u[16];
  __bf16 h[16];
};

// Native bf16 convert (RNE) -> v_cvt_pk_bf16_f32 when paired.
static __device__ __forceinline__ unsigned short f2bf(float f) {
  union { __bf16 h; unsigned short s; } cv;
  cv.h = (__bf16)f;
  return cv.s;
}

static __device__ __forceinline__ f32x8 wmma_bf16(const Frag& a, const Frag& b, f32x8 c) {
  return __builtin_amdgcn_wmma_f32_16x16x32_bf16(false, a.bf, false, b.bf,
                                                 (short)0, c, false, false);
}

// Load a 16x32 bf16 fragment from row-major [rows, ld] memory.
// ISA 7.12.2: lane l<16 -> row l, K {0..7,16..23}; lane l+16 -> K {8..15,24..31}.
static __device__ __forceinline__ void load_frag(Frag& fr, const unsigned short* base,
                                                 int row0, int ld, int kk,
                                                 int lrow, int hi) {
  const unsigned short* p = base + (size_t)(row0 + lrow) * ld + kk + hi * 8;
  fr.q[0] = *(const uint4*)p;
  fr.q[1] = *(const uint4*)(p + 16);
}

// ---------------------------------------------------------------------------
// Sinusoid position table (pos_q == pos_c): f32 + bf16 copies.
__global__ void k_pos(float* __restrict__ pos_f, unsigned short* __restrict__ pos_bf) {
  int idx = blockIdx.x * 256 + threadIdx.x;
  if (idx >= 256 * 192) return;
  int p = idx / 192, j = idx % 192;
  float ex = (float)(2 * (j / 2)) * (1.0f / 192.0f);
  float angle = (float)p / __powf(10000.0f, ex);
  float v = (j & 1) ? __cosf(angle) : __sinf(angle);
  pos_f[idx] = v;
  pos_bf[idx] = f2bf(v);
}

// f32 -> bf16 tensor conversion (weights).
__global__ void k_f2bf(const float* __restrict__ in, unsigned short* __restrict__ out, int n) {
  int i = blockIdx.x * 256 + threadIdx.x;
  if (i < n) out[i] = f2bf(in[i]);
}

// ---------------------------------------------------------------------------
// Patch embed: out[b,p,:] = patch(x) @ W^T + bias + pos.  Patch extraction is
// fused into the A-fragment loader (8-float runs of one image row -> bf16).
static __device__ __forceinline__ void load_a_patch(Frag& a, const float* __restrict__ x,
                                                    int b, int pr, int pc,
                                                    int kk, int hi) {
#pragma unroll
  for (int ro = 0; ro < 2; ++ro) {
    int kb  = kk + hi * 8 + ro * 16;        // K = c*256 + py*16 + px
    int c   = kb >> 8;
    int py  = (kb >> 4) & 15;
    int px0 = kb & 15;                      // 0 or 8 -> 8-float aligned run
    const float* src = x + (((size_t)(b * 3 + c) * 256 + pr * 16 + py) * 256
                            + pc * 16 + px0);
    float4 f0 = *(const float4*)src;
    float4 f1 = *(const float4*)(src + 4);
    a.h[ro * 8 + 0] = (__bf16)f0.x; a.h[ro * 8 + 1] = (__bf16)f0.y;
    a.h[ro * 8 + 2] = (__bf16)f0.z; a.h[ro * 8 + 3] = (__bf16)f0.w;
    a.h[ro * 8 + 4] = (__bf16)f1.x; a.h[ro * 8 + 5] = (__bf16)f1.y;
    a.h[ro * 8 + 6] = (__bf16)f1.z; a.h[ro * 8 + 7] = (__bf16)f1.w;
  }
}

// Wave = 16 patches x 64 embed cols (1 M-tile x 4 N-tiles), double-buffered.
__global__ void k_embed(const float* __restrict__ x,
                        const unsigned short* __restrict__ Wbf,   // [192,768] bf16
                        const float* __restrict__ bias,           // [192]
                        const float* __restrict__ pos,            // [256,192] f32
                        unsigned short* __restrict__ out) {       // [B*256,192] bf16
  int lane = threadIdx.x & 31, wid = threadIdx.x >> 5;
  int lrow = lane & 15, hi = lane >> 4;
  int wg = blockIdx.x * 8 + wid;            // 64*16*3 = 3072 waves
  int ng = wg % 3;
  int mt = (wg / 3) & 15;
  int b  = wg / 48;
  int p  = mt * 16 + lrow;                  // A row = patch index in batch
  int pr = p >> 4, pc = p & 15;

  f32x8 acc[4] = {};
  Frag a[2], bf[2][4];
  load_a_patch(a[0], x, b, pr, pc, 0, hi);
#pragma unroll
  for (int t = 0; t < 4; ++t) load_frag(bf[0][t], Wbf, ng * 64 + t * 16, 768, 0, lrow, hi);

#pragma unroll
  for (int i = 0; i < 24; ++i) {            // K = 768 = 24 x 32
    const int cur = i & 1, nxt = cur ^ 1;
    if (i + 1 < 24) {
      load_a_patch(a[nxt], x, b, pr, pc, (i + 1) * 32, hi);
#pragma unroll
      for (int t = 0; t < 4; ++t)
        load_frag(bf[nxt][t], Wbf, ng * 64 + t * 16, 768, (i + 1) * 32, lrow, hi);
    }
#pragma unroll
    for (int t = 0; t < 4; ++t) acc[t] = wmma_bf16(a[cur], bf[cur][t], acc[t]);
  }

#pragma unroll
  for (int t = 0; t < 4; ++t) {
    int n = ng * 64 + t * 16 + lrow;
    float bn = bias[n];
#pragma unroll
    for (int r = 0; r < 8; ++r) {
      int pm = mt * 16 + hi * 8 + r;
      float v = acc[t][r] + bn + pos[pm * 192 + n];
      out[((size_t)b * 256 + pm) * 192 + n] = f2bf(v);
    }
  }
}

// ---------------------------------------------------------------------------
// Generic GEMM: C[M,N] = A[M,K] * B[N,K]^T, each wave = 32 rows x 64 cols
// (2 M-tiles x 4 N-tiles).  A double-buffered across K-steps; B loaded per
// N-tile and immediately consumed by TWO WMMAs (wait amortized).
// EP 0: bf16 row-major store. EP 1: bf16 transposed store (contiguous uint4
//       per tile row -> used to produce V^T). EP 2: f32 + bias store.
template <int EP, int K>
__global__ void k_gemm(const unsigned short* __restrict__ A, int lda,
                       const unsigned short* __restrict__ B, int ldb,
                       void* __restrict__ C, int ldc,
                       const float* __restrict__ bias,
                       int Mpairs, int Ntiles) {
  int lane = threadIdx.x & 31, wid = threadIdx.x >> 5;
  int lrow = lane & 15, hi = lane >> 4;
  int wg = blockIdx.x * (blockDim.x >> 5) + wid;
  int ngroups = Ntiles >> 2;
  int mp = wg / ngroups, ng = wg % ngroups;
  if (mp >= Mpairs) return;

  constexpr int NK = K / 32;
  f32x8 acc[2][4] = {};
  Frag a[2][2];
  load_frag(a[0][0], A, mp * 32,      lda, 0, lrow, hi);
  load_frag(a[0][1], A, mp * 32 + 16, lda, 0, lrow, hi);

#pragma unroll
  for (int i = 0; i < NK; ++i) {
    const int cur = i & 1, nxt = cur ^ 1;
    if (i + 1 < NK) {
      load_frag(a[nxt][0], A, mp * 32,      lda, (i + 1) * 32, lrow, hi);
      load_frag(a[nxt][1], A, mp * 32 + 16, lda, (i + 1) * 32, lrow, hi);
    }
#pragma unroll
    for (int t = 0; t < 4; ++t) {
      Frag b;
      load_frag(b, B, ng * 64 + t * 16, ldb, i * 32, lrow, hi);
      acc[0][t] = wmma_bf16(a[cur][0], b, acc[0][t]);
      acc[1][t] = wmma_bf16(a[cur][1], b, acc[1][t]);
    }
  }

#pragma unroll
  for (int m = 0; m < 2; ++m) {
    int rowb = mp * 32 + m * 16;
#pragma unroll
    for (int t = 0; t < 4; ++t) {
      int n = ng * 64 + t * 16 + lrow;
      if (EP == 0) {
        unsigned short* Cb = (unsigned short*)C;
#pragma unroll
        for (int r = 0; r < 8; ++r)
          Cb[(size_t)(rowb + hi * 8 + r) * ldc + n] = f2bf(acc[m][t][r]);
      } else if (EP == 1) {
        union { uint4 q; unsigned short u[8]; } pk;
#pragma unroll
        for (int r = 0; r < 8; ++r) pk.u[r] = f2bf(acc[m][t][r]);
        unsigned short* Cb = (unsigned short*)C;
        *(uint4*)(Cb + (size_t)n * ldc + rowb + hi * 8) = pk.q;
      } else {
        float* Cf = (float*)C;
        float bn = bias[n];
#pragma unroll
        for (int r = 0; r < 8; ++r)
          Cf[(size_t)(rowb + hi * 8 + r) * ldc + n] = acc[m][t][r] + bn;
      }
    }
  }
}

// ---------------------------------------------------------------------------
// Fused attention: one wave per (b, h, 16-query tile). QK^T (16 WMMA, K=32),
// softmax in LDS, AV (16 WMMA, K=256) against transposed V.
__global__ void __launch_bounds__(32)
k_attn(const unsigned short* __restrict__ q,     // [B*256,192] bf16
       const unsigned short* __restrict__ kmat,  // [256,192]   bf16
       const unsigned short* __restrict__ vt,    // [192,B*256] bf16 (V^T)
       float* __restrict__ out) {                // [B*256,192] f32
  __shared__ float          s_sim[16 * 256];     // 16 KB logits
  __shared__ unsigned short s_attn[16 * 256];    //  8 KB bf16 attn
  int lane = threadIdx.x;
  int lrow = lane & 15, hi = lane >> 4;
  int mt = blockIdx.x & 15;
  int h  = (blockIdx.x >> 4) % 6;
  int b  = blockIdx.x / 96;
  const float scale = 0.17677669529663687f;      // 32^-0.5

  // Phase 1: S = q . k^T (A reused across all 16 key tiles; keys double-buffered)
  Frag a;
  load_frag(a, q, b * 256 + mt * 16, 192, h * 32, lrow, hi);
  Frag kb[2];
  load_frag(kb[0], kmat, 0, 192, h * 32, lrow, hi);
#pragma unroll
  for (int jt = 0; jt < 16; ++jt) {
    const int cur = jt & 1, nxt = cur ^ 1;
    if (jt + 1 < 16)
      load_frag(kb[nxt], kmat, (jt + 1) * 16, 192, h * 32, lrow, hi);
    f32x8 s = {};
    s = wmma_bf16(a, kb[cur], s);
#pragma unroll
    for (int r = 0; r < 8; ++r)
      s_sim[(hi * 8 + r) * 256 + jt * 16 + lrow] = s[r] * scale;
  }
  __syncthreads();

  // Phase 2: softmax; lanes l and l+16 split row l over 128-col halves.
  {
    float* rowp = s_sim + lrow * 256 + hi * 128;
    float m = -3.4e38f;
    for (int i = 0; i < 128; ++i) m = fmaxf(m, rowp[i]);
    m = fmaxf(m, __shfl_xor(m, 16, 32));
    float sum = 0.f;
    for (int i = 0; i < 128; ++i) { float e = __expf(rowp[i] - m); rowp[i] = e; sum += e; }
    sum += __shfl_xor(sum, 16, 32);
    float inv = 1.0f / sum;
    unsigned short* ap = s_attn + lrow * 256 + hi * 128;
    for (int i = 0; i < 128; ++i) ap[i] = f2bf(rowp[i] * inv);
  }
  __syncthreads();

  // Phase 3: out = attn . v  (K=256 over 8 steps, N=32 as 2 tiles)
  f32x8 acc[2] = {};
  Frag at[2], vb[2][2];
  {
    const unsigned short* p = s_attn + lrow * 256 + hi * 8;
    at[0].q[0] = *(const uint4*)p;
    at[0].q[1] = *(const uint4*)(p + 16);
  }
#pragma unroll
  for (int nt = 0; nt < 2; ++nt)
    load_frag(vb[0][nt], vt, h * 32 + nt * 16, 16384, b * 256, lrow, hi);

#pragma unroll
  for (int ks = 0; ks < 8; ++ks) {
    const int cur = ks & 1, nxt = cur ^ 1;
    if (ks + 1 < 8) {
      const unsigned short* p = s_attn + lrow * 256 + (ks + 1) * 32 + hi * 8;
      at[nxt].q[0] = *(const uint4*)p;
      at[nxt].q[1] = *(const uint4*)(p + 16);
#pragma unroll
      for (int nt = 0; nt < 2; ++nt)
        load_frag(vb[nxt][nt], vt, h * 32 + nt * 16, 16384,
                  b * 256 + (ks + 1) * 32, lrow, hi);
    }
#pragma unroll
    for (int nt = 0; nt < 2; ++nt)
      acc[nt] = wmma_bf16(at[cur], vb[cur][nt], acc[nt]);
  }
#pragma unroll
  for (int nt = 0; nt < 2; ++nt) {
    int d = h * 32 + nt * 16 + lrow;
#pragma unroll
    for (int r = 0; r < 8; ++r)
      out[((size_t)b * 256 + mt * 16 + hi * 8 + r) * 192 + d] = acc[nt][r];
  }
}

// ---------------------------------------------------------------------------
// 2x bilinear upsample (align_corners=True) of 16x16 token grid -> 32x32,
// emitting the [B*1024,192] bf16 A-matrix for the output projection.
__global__ void k_upsample(const float* __restrict__ ao,          // [B*256,192]
                           unsigned short* __restrict__ flat) {   // [B*1024,192]
  int idx = blockIdx.x * 256 + threadIdx.x;
  if (idx >= 64 * 1024 * 192) return;
  int e = idx % 192;
  int t = (idx / 192) & 1023;
  int b = idx / (1024 * 192);
  int oy = t >> 5, ox = t & 31;
  float sy = oy * (15.0f / 31.0f);
  float sx = ox * (15.0f / 31.0f);
  int i0 = (int)sy; float wy = sy - (float)i0; int i1 = min(i0 + 1, 15);
  int j0 = (int)sx; float wx = sx - (float)j0; int j1 = min(j0 + 1, 15);
  const float* base = ao + (size_t)b * 256 * 192 + e;
  float v00 = base[(i0 * 16 + j0) * 192];
  float v01 = base[(i0 * 16 + j1) * 192];
  float v10 = base[(i1 * 16 + j0) * 192];
  float v11 = base[(i1 * 16 + j1) * 192];
  float v = (1.f - wy) * ((1.f - wx) * v00 + wx * v01)
          +         wy * ((1.f - wx) * v10 + wx * v11);
  flat[idx] = f2bf(v);
}

// ---------------------------------------------------------------------------
extern "C" void kernel_launch(void* const* d_in, const int* in_sizes, int n_in,
                              void* d_out, int out_size, void* d_ws, size_t ws_size,
                              hipStream_t stream) {
  (void)in_sizes; (void)n_in; (void)out_size; (void)ws_size;
  const float* x        = (const float*)d_in[0];
  const float* ctx      = (const float*)d_in[1];
  const float* wq_patch = (const float*)d_in[2];
  const float* bq       = (const float*)d_in[3];
  const float* wc_patch = (const float*)d_in[4];
  const float* bc       = (const float*)d_in[5];
  const float* Wq       = (const float*)d_in[6];
  const float* Wk       = (const float*)d_in[7];
  const float* Wv       = (const float*)d_in[8];
  const float* Wo       = (const float*)d_in[9];
  const float* bo       = (const float*)d_in[10];

  char* ws = (char*)d_ws;
  size_t off = 0;
  auto alloc = [&](size_t bytes) -> void* {
    void* p = ws + off;
    off = (off + bytes + 255) & ~(size_t)255;
    return p;
  };
  float*          pos_f  = (float*)         alloc(49152u * 4);   // [256,192]
  unsigned short* pos_bf = (unsigned short*)alloc(49152u * 2);
  unsigned short* Wembq  = (unsigned short*)alloc(147456u * 2);  // [192,768]
  unsigned short* Wembc  = (unsigned short*)alloc(147456u * 2);
  unsigned short* Wq_bf  = (unsigned short*)alloc(36864u * 2);
  unsigned short* Wk_bf  = (unsigned short*)alloc(36864u * 2);
  unsigned short* Wv_bf  = (unsigned short*)alloc(36864u * 2);
  unsigned short* Wo_bf  = (unsigned short*)alloc(24576u * 2);
  unsigned short* xq_bf  = (unsigned short*)alloc(3145728u * 2); // [16384,192]
  unsigned short* ct_bf  = (unsigned short*)alloc(3145728u * 2);
  unsigned short* q_bf   = (unsigned short*)alloc(3145728u * 2);
  unsigned short* k_bf   = (unsigned short*)alloc(49152u * 2);   // [256,192]
  unsigned short* vt_bf  = (unsigned short*)alloc(3145728u * 2); // [192,16384] V^T
  float*          at_out = (float*)         alloc(3145728u * 4); // [16384,192]
  unsigned short* fl_bf  = (unsigned short*)alloc(12582912u * 2);// [65536,192]

  k_pos<<<192, 256, 0, stream>>>(pos_f, pos_bf);
  k_f2bf<<<576, 256, 0, stream>>>(wq_patch, Wembq, 147456);
  k_f2bf<<<576, 256, 0, stream>>>(wc_patch, Wembc, 147456);
  k_f2bf<<<144, 256, 0, stream>>>(Wq, Wq_bf, 36864);
  k_f2bf<<<144, 256, 0, stream>>>(Wk, Wk_bf, 36864);
  k_f2bf<<<144, 256, 0, stream>>>(Wv, Wv_bf, 36864);
  k_f2bf<<<96,  256, 0, stream>>>(Wo, Wo_bf, 24576);

  k_embed<<<384, 256, 0, stream>>>(x,   Wembq, bq, pos_f, xq_bf);
  k_embed<<<384, 256, 0, stream>>>(ctx, Wembc, bc, pos_f, ct_bf);

  // q = xq @ Wq^T ; k = pos @ Wk^T (batch-independent!) ; v^T from ctx @ Wv^T
  k_gemm<0, 192><<<192, 256, 0, stream>>>(xq_bf, 192, Wq_bf, 192, q_bf, 192, nullptr, 512, 12);
  k_gemm<0, 192><<<3,   256, 0, stream>>>(pos_bf, 192, Wk_bf, 192, k_bf, 192, nullptr, 8, 12);
  k_gemm<1, 192><<<192, 256, 0, stream>>>(ct_bf, 192, Wv_bf, 192, vt_bf, 16384, nullptr, 512, 12);

  k_attn<<<6144, 32, 0, stream>>>(q_bf, k_bf, vt_bf, at_out);
  k_upsample<<<49152, 256, 0, stream>>>(at_out, fl_bf);

  // out = flat @ Wo^T + bo -> [65536,128] f32
  k_gemm<2, 192><<<512, 256, 0, stream>>>(fl_bf, 192, Wo_bf, 192, d_out, 128, bo, 2048, 8);
}
